// _HFRefAttention_19000935317691
// MI455X (gfx1250) — compile-verified, matrix operand order checked by reading, not run on hardware
//
#include <hip/hip_runtime.h>
#include <hip/hip_bf16.h>

// ---------------------------------------------------------------------------
// Types for CDNA5 WMMA (wave32): bf16 16x16x32, f32 accumulate.
// ---------------------------------------------------------------------------
typedef __attribute__((ext_vector_type(16))) __bf16 v16bf;
typedef __attribute__((ext_vector_type(8)))  __bf16 v8bf;
typedef __attribute__((ext_vector_type(8)))  float  v8f;

#define S_LEN   2048
#define HIDDEN_ 2048
#define Q_RANK_ 1536
#define KV_RANK_ 512
#define D_NOPE_ 128
#define D_ROPE_ 64
#define D_QK_   192
#define D_V_    128
#define NHEAD_  16

// --------------------------- TDM availability ------------------------------
#if defined(__has_builtin)
#  if __has_builtin(__builtin_amdgcn_tensor_load_to_lds) && \
      __has_builtin(__builtin_amdgcn_s_wait_tensorcnt)
#    define HAVE_TDM 1
#  endif
#endif
#ifndef HAVE_TDM
#  define HAVE_TDM 0
#endif

#if HAVE_TDM
typedef __attribute__((ext_vector_type(4))) unsigned v4u;
typedef __attribute__((ext_vector_type(8))) int      v8i;
typedef __attribute__((ext_vector_type(4))) int      v4i;

// Issue a 2D TDM tile load (bf16 elements) global -> LDS.
// Descriptor layout per CDNA5 ISA (groups 0/1; 2D, so
// groups 2/3 are zero). All dims/strides in element (2-byte) units.
// This toolchain exposes the 6-arg builtin:
//   (uint32x4 g0, int32x8 g1, int32x4 g2, int32x4 g3, int32x8 pad, i32 cpol)
__device__ __forceinline__ void tdm_load_2d(unsigned lds_off, const void* gtile,
                                            unsigned tdim0, unsigned tdim1,
                                            unsigned tile0, unsigned tile1,
                                            unsigned stride0) {
  unsigned long long ga = (unsigned long long)(uintptr_t)gtile;
  v4u g0;
  g0[0] = 1u;                                  // count=1 (valid user D#)
  g0[1] = lds_off;                             // lds_addr (bytes)
  g0[2] = (unsigned)ga;                        // global_addr[31:0]
  g0[3] = (unsigned)((ga >> 32) & 0x01FFFFFFu) // global_addr[56:32]
          | (2u << 30);                        // type = 2 ("image")
  v8i g1;
  g1[0] = (int)(1u << 16);                     // workgroup_mask=0, data_size=1 (2B)
  g1[1] = (int)((tdim0 & 0xFFFFu) << 16);      // tensor_dim0 (bits 79:48)
  g1[2] = (int)((tdim0 >> 16) | ((tdim1 & 0xFFFFu) << 16));  // tensor_dim1 (111:80)
  g1[3] = (int)((tdim1 >> 16) | (tile0 << 16));              // tile_dim0 (127:112)
  g1[4] = (int)(tile1 & 0xFFFFu);              // tile_dim1 (143:128), tile_dim2=0
  g1[5] = (int)stride0;                        // tensor_dim0_stride (207:160)
  g1[6] = 0;
  g1[7] = 0;
  v4i z4 = {0, 0, 0, 0};
  v8i z8 = {0, 0, 0, 0, 0, 0, 0, 0};
  __builtin_amdgcn_tensor_load_to_lds(g0, g1, z4, z4, z8, 0);
}
#endif

__device__ __forceinline__ v8f wmma_bf16(v16bf a, v16bf b, v8f c) {
  return __builtin_amdgcn_wmma_f32_16x16x32_bf16(false, a, false, b, (short)0, c,
                                                 false, false);
}

// Load a 16x32 bf16 A/B fragment from row-major memory (K contiguous).
// ISA layout: lanes 0-15 hold row (lane), K = {0..7, 16..23};
//             lanes 16-31 hold row (lane-16), K = {8..15, 24..31}.
__device__ __forceinline__ v16bf load_frag(const __bf16* base, int ld, int lane) {
  int row = lane & 15;
  int ko  = (lane >> 4) * 8;
  const __bf16* p = base + row * ld + ko;
  v8bf lo = *(const v8bf*)p;         // K = ko .. ko+7
  v8bf hi = *(const v8bf*)(p + 16);  // K = ko+16 .. ko+23
  v16bf f;
#pragma unroll
  for (int i = 0; i < 8; ++i) { f[i] = lo[i]; f[i + 8] = hi[i]; }
  return f;
}

// ---------------------------------------------------------------------------
// fp32 -> bf16 convert
// ---------------------------------------------------------------------------
__global__ void cvt_f32_bf16(const float* __restrict__ x, __bf16* __restrict__ y,
                             int n) {
  int i = blockIdx.x * blockDim.x + threadIdx.x;
  if (i < n) y[i] = (__bf16)x[i];
}

// ---------------------------------------------------------------------------
// Row RMSNorm (over L leading elements of a row with given stride) -> bf16
// ---------------------------------------------------------------------------
__global__ __launch_bounds__(256) void rmsnorm_bf16(const float* __restrict__ x,
                                                    const float* __restrict__ w,
                                                    __bf16* __restrict__ y,
                                                    int stride, int L) {
  int row = blockIdx.x;
  const float* xr = x + (size_t)row * stride;
  float ss = 0.f;
  for (int i = threadIdx.x; i < L; i += 256) { float v = xr[i]; ss += v * v; }
#pragma unroll
  for (int m = 16; m >= 1; m >>= 1) ss += __shfl_xor(ss, m, 32);
  __shared__ float red[8];
  int wv = threadIdx.x >> 5, ln = threadIdx.x & 31;
  if (ln == 0) red[wv] = ss;
  __syncthreads();
  float tot = 0.f;
#pragma unroll
  for (int j = 0; j < 8; ++j) tot += red[j];
  float inv = rsqrtf(tot / (float)L + 1e-6f);
  for (int i = threadIdx.x; i < L; i += 256)
    y[(size_t)row * L + i] = (__bf16)(w[i] * xr[i] * inv);
}

// ---------------------------------------------------------------------------
// Generic bf16 TN GEMM: C[M,N](f32) = A[M,K] * B[N,K]^T.
// 64x64 block tile, 4 waves (128 threads), each wave -> 32x32 via 2x2 WMMA.
// M,N multiples of 64; K multiple of 32.
// ---------------------------------------------------------------------------
__global__ __launch_bounds__(128) void gemm_bf16_tn(const __bf16* __restrict__ A,
                                                    const __bf16* __restrict__ B,
                                                    float* __restrict__ C,
                                                    int M, int N, int K) {
  __shared__ __align__(16) __bf16 As[64][48];  // 32 + 16 pad (96B stride, 16B aligned)
  __shared__ __align__(16) __bf16 Bs[64][48];
  int tid  = threadIdx.x;
  int wave = tid >> 5, lane = tid & 31;
  int bm = blockIdx.x * 64, bn = blockIdx.y * 64;
  int wm = (wave >> 1) * 32, wn = (wave & 1) * 32;

  v8f acc00 = {}, acc01 = {}, acc10 = {}, acc11 = {};

  for (int k0 = 0; k0 < K; k0 += 32) {
#pragma unroll
    for (int it = 0; it < 2; ++it) {
      int idx = tid + it * 128;             // 0..255: 64 rows x 4 8-elt chunks
      int r = idx >> 2, c = (idx & 3) * 8;
      *(v8bf*)&As[r][c] = *(const v8bf*)(A + (size_t)(bm + r) * K + k0 + c);
      *(v8bf*)&Bs[r][c] = *(const v8bf*)(B + (size_t)(bn + r) * K + k0 + c);
    }
    // L2 prefetch of next K slab (global_prefetch_b8)
    if (k0 + 32 < K) {
      __builtin_prefetch(A + (size_t)(bm + (tid >> 1)) * K + k0 + 32 + (tid & 1) * 16,
                         0, 1);
      __builtin_prefetch(B + (size_t)(bn + (tid >> 1)) * K + k0 + 32 + (tid & 1) * 16,
                         0, 1);
    }
    __syncthreads();
    v16bf a0 = load_frag(&As[wm][0],      48, lane);
    v16bf a1 = load_frag(&As[wm + 16][0], 48, lane);
    v16bf b0 = load_frag(&Bs[wn][0],      48, lane);
    v16bf b1 = load_frag(&Bs[wn + 16][0], 48, lane);
    acc00 = wmma_bf16(a0, b0, acc00);
    acc01 = wmma_bf16(a0, b1, acc01);
    acc10 = wmma_bf16(a1, b0, acc10);
    acc11 = wmma_bf16(a1, b1, acc11);
    __syncthreads();
  }

  int lr = (lane >> 4) * 8, lc = lane & 15;
#pragma unroll
  for (int r = 0; r < 8; ++r) {
    C[(size_t)(bm + wm +      lr + r) * N + bn + wn +      lc] = acc00[r];
    C[(size_t)(bm + wm +      lr + r) * N + bn + wn + 16 + lc] = acc01[r];
    C[(size_t)(bm + wm + 16 + lr + r) * N + bn + wn +      lc] = acc10[r];
    C[(size_t)(bm + wm + 16 + lr + r) * N + bn + wn + 16 + lc] = acc11[r];
  }
}

// ---------------------------------------------------------------------------
// RoPE + repack: builds per-head q_full [H][S][192], k_full [H][S][192],
// and transposed V  vT [H][128][S], all bf16.
// grid (S, H), 128 threads.
// ---------------------------------------------------------------------------
__global__ __launch_bounds__(128) void rope_pack(const float* __restrict__ q,    // [S][H*192]
                                                 const float* __restrict__ kvb,  // [S][H*256]
                                                 const float* __restrict__ kva,  // [S][576]
                                                 const int*   __restrict__ pos,  // [S]
                                                 __bf16* __restrict__ qfull,
                                                 __bf16* __restrict__ kfull,
                                                 __bf16* __restrict__ vT) {
  int t = blockIdx.x, h = blockIdx.y, tid = threadIdx.x;
  float p = (float)pos[t];
  const float* qrow  = q   + ((size_t)t * NHEAD_ + h) * D_QK_;
  const float* kvrow = kvb + ((size_t)t * NHEAD_ + h) * (D_NOPE_ + D_V_);
  __bf16* qf = qfull + ((size_t)h * S_LEN + t) * D_QK_;
  __bf16* kf = kfull + ((size_t)h * S_LEN + t) * D_QK_;

  int d = tid;  // 0..127
  qf[d] = (__bf16)qrow[d];                                   // q_nope
  kf[d] = (__bf16)kvrow[d];                                  // k_nope
  vT[((size_t)h * D_V_ + d) * S_LEN + t] = (__bf16)kvrow[D_NOPE_ + d];  // v^T

  if (tid < D_ROPE_) {
    int j = d & 31;
    float freq = __powf(10000.0f, -(float)(2 * j) / (float)D_ROPE_);
    float ang = p * freq;
    float c = __cosf(ang), sn = __sinf(ang);
    float x  = qrow[D_NOPE_ + d];
    float xo = (d < 32) ? -qrow[D_NOPE_ + d + 32] : qrow[D_NOPE_ + d - 32];
    qf[D_NOPE_ + d] = (__bf16)(x * c + xo * sn);
    const float* kpe = kva + (size_t)t * (KV_RANK_ + D_ROPE_) + KV_RANK_;
    float kx  = kpe[d];
    float kxo = (d < 32) ? -kpe[d + 32] : kpe[d - 32];
    kf[D_NOPE_ + d] = (__bf16)(kx * c + kxo * sn);
  }
}

// ---------------------------------------------------------------------------
// Flash attention (causal, per head). grid (S/64, H), 128 threads (4 waves).
// Wave w owns 16 query rows; Q fragments resident in VGPRs. K tile and V^T
// tile staged in LDS via the Tensor Data Mover (TDM) when available (wave 0
// issues tensor_load_to_lds, waits TENSORcnt, then a workgroup barrier
// publishes the tiles); otherwise cooperative vector loads. P is transposed
// from C-layout to A-layout through a per-wave LDS scratch.
// Output written bf16 as attn[S][H*128] (feeds the o_proj GEMM directly).
// ---------------------------------------------------------------------------
#if HAVE_TDM
#define KS_LD 192   // TDM writes packed rows (no LDS padding); 384B, 16B aligned
#define VS_LD 64    // 128B rows
#else
#define KS_LD 208
#define VS_LD 80
#endif

__global__ __launch_bounds__(128) void mla_flash(const __bf16* __restrict__ Qf,
                                                 const __bf16* __restrict__ Kf,
                                                 const __bf16* __restrict__ Vt,
                                                 __bf16* __restrict__ O) {
  const float scale = 0.07216878364870323f;  // 192^-0.5
  int h  = blockIdx.y;
  int q0 = blockIdx.x * 64;
  int wave = threadIdx.x >> 5;
  int lane = threadIdx.x & 31;
  int halfsel = lane >> 4;
  int lcol = lane & 15;

  const __bf16* Qh = Qf + (size_t)h * S_LEN * D_QK_;
  const __bf16* Kh = Kf + (size_t)h * S_LEN * D_QK_;
  const __bf16* Vh = Vt + (size_t)h * D_V_ * S_LEN;

  __shared__ __align__(16) __bf16 Ks[64][KS_LD];   // 64 keys x 192
  __shared__ __align__(16) __bf16 Vs[128][VS_LD];  // 128 d x 64 keys
  __shared__ __align__(16) __bf16 Ps[4][16][48];   // per-wave P scratch 16x32 (+16)

  int qr = q0 + wave * 16;  // wave's first query row

  // Q fragments: 6 K-chunks of 16x32, straight from global (row-major, 16B aligned)
  v16bf qfrag[6];
#pragma unroll
  for (int kc = 0; kc < 6; ++kc)
    qfrag[kc] = load_frag(Qh + (size_t)qr * D_QK_ + kc * 32, D_QK_, lane);

  v8f o_acc[8];
#pragma unroll
  for (int i = 0; i < 8; ++i) {
    v8f z = {};
    o_acc[i] = z;
  }
  float mrow[8], lrow[8];
#pragma unroll
  for (int r = 0; r < 8; ++r) { mrow[r] = -3.0e38f; lrow[r] = 0.f; }

  int nkb = blockIdx.x + 1;  // causal: key blocks 0..blockIdx.x
  for (int kb = 0; kb < nkb; ++kb) {
#if HAVE_TDM
    if (threadIdx.x < 32) {  // wave 0 drives the Tensor Data Mover
      // K tile: 64 rows x 192 cols of K_full[h], row stride 192 elements
      tdm_load_2d((unsigned)(uintptr_t)(void*)&Ks[0][0],
                  Kh + (size_t)(kb * 64) * D_QK_,
                  D_QK_, S_LEN, D_QK_, 64, D_QK_);
      // V^T tile: 128 rows x 64 cols of vT[h], row stride 2048 elements
      tdm_load_2d((unsigned)(uintptr_t)(void*)&Vs[0][0],
                  Vh + kb * 64,
                  S_LEN, D_V_, 64, D_V_, S_LEN);
      __builtin_amdgcn_s_wait_tensorcnt(0);
    }
    __syncthreads();
#else
    // Stage K tile 64x192 (1536 16B-chunks / 128 threads = 12 each)
#pragma unroll
    for (int it = 0; it < 12; ++it) {
      int idx = threadIdx.x + it * 128;
      int r = idx / 24, c = (idx % 24) * 8;
      *(v8bf*)&Ks[r][c] = *(const v8bf*)(Kh + (size_t)(kb * 64 + r) * D_QK_ + c);
    }
    // Stage V^T tile 128x64
#pragma unroll
    for (int it = 0; it < 8; ++it) {
      int idx = threadIdx.x + it * 128;
      int dd = idx >> 3, c = (idx & 7) * 8;
      *(v8bf*)&Vs[dd][c] = *(const v8bf*)(Vh + (size_t)dd * S_LEN + kb * 64 + c);
    }
    __syncthreads();
#endif

    for (int pair = 0; pair < 2; ++pair) {          // two 32-key chunks
      if (kb * 64 + pair * 32 > qr + 15) break;     // wave-uniform causal skip
#pragma unroll
      for (int sub = 0; sub < 2; ++sub) {           // two 16-key subtiles
        int kt = pair * 2 + sub;
        int kbase = kb * 64 + kt * 16;
        if (kbase > qr + 15) {                      // fully masked: zero P half
#pragma unroll
          for (int r = 0; r < 8; ++r)
            Ps[wave][r + 8 * halfsel][sub * 16 + lcol] = (__bf16)0.f;
          continue;
        }
        // S tile = Q(16x192) . K^T(192x16)
        v8f sacc = {};
#pragma unroll
        for (int kc = 0; kc < 6; ++kc)
          sacc = wmma_bf16(qfrag[kc],
                           load_frag(&Ks[kt * 16][kc * 32], KS_LD, lane), sacc);

        float corr[8];
#pragma unroll
        for (int r = 0; r < 8; ++r) {
          int rowg = qr + r + 8 * halfsel;
          float sv = sacc[r] * scale;
          if (kbase + lcol > rowg) sv = -3.0e38f;   // causal mask
          // row max across 16-lane half
          float mx = sv;
          mx = fmaxf(mx, __shfl_xor(mx, 1, 32));
          mx = fmaxf(mx, __shfl_xor(mx, 2, 32));
          mx = fmaxf(mx, __shfl_xor(mx, 4, 32));
          mx = fmaxf(mx, __shfl_xor(mx, 8, 32));
          float mnew = fmaxf(mrow[r], mx);
          float cr = __expf(mrow[r] - mnew);
          float pv = __expf(sv - mnew);
          float ps = pv;
          ps += __shfl_xor(ps, 1, 32);
          ps += __shfl_xor(ps, 2, 32);
          ps += __shfl_xor(ps, 4, 32);
          ps += __shfl_xor(ps, 8, 32);
          lrow[r] = lrow[r] * cr + ps;
          mrow[r] = mnew;
          corr[r] = cr;
          Ps[wave][r + 8 * halfsel][sub * 16 + lcol] = (__bf16)pv;
        }
        // rescale running O by exp(m_old - m_new)
#pragma unroll
        for (int nt = 0; nt < 8; ++nt)
#pragma unroll
          for (int r = 0; r < 8; ++r) o_acc[nt][r] *= corr[r];
      }
      // P(16x32) . V(32x128): 8 WMMA along d
      asm volatile("s_wait_dscnt 0x0" ::: "memory");
      v16bf pfrag = load_frag(&Ps[wave][0][0], 48, lane);
#pragma unroll
      for (int nt = 0; nt < 8; ++nt) {
        v16bf vfrag = load_frag(&Vs[nt * 16][pair * 32], VS_LD, lane);
        o_acc[nt] = wmma_bf16(pfrag, vfrag, o_acc[nt]);
      }
    }
    __syncthreads();
  }

  // epilogue: O / l, write bf16 attn [S][H*128]
#pragma unroll
  for (int r = 0; r < 8; ++r) lrow[r] = 1.f / lrow[r];
#pragma unroll
  for (int nt = 0; nt < 8; ++nt)
#pragma unroll
    for (int r = 0; r < 8; ++r) {
      int rowg = qr + r + 8 * halfsel;
      int dd = nt * 16 + lcol;
      O[(size_t)rowg * (NHEAD_ * D_V_) + h * D_V_ + dd] =
          (__bf16)(o_acc[nt][r] * lrow[r]);
    }
}

// ---------------------------------------------------------------------------
// Launch
// ---------------------------------------------------------------------------
extern "C" void kernel_launch(void* const* d_in, const int* in_sizes, int n_in,
                              void* d_out, int out_size, void* d_ws, size_t ws_size,
                              hipStream_t stream) {
  (void)in_sizes; (void)n_in; (void)out_size; (void)ws_size;
  const float* hidden  = (const float*)d_in[0];
  const int*   pos     = (const int*)d_in[1];
  const float* q_a_w   = (const float*)d_in[2];
  const float* q_a_ln  = (const float*)d_in[3];
  const float* q_b_w   = (const float*)d_in[4];
  const float* kv_a_w  = (const float*)d_in[5];
  const float* kv_a_ln = (const float*)d_in[6];
  const float* kv_b_w  = (const float*)d_in[7];
  const float* o_w     = (const float*)d_in[8];
  float* out = (float*)d_out;

  char* ws = (char*)d_ws;
  size_t off = 0;
  auto alloc = [&](size_t bytes) -> void* {
    void* p = ws + off;
    off += (bytes + 255) & ~(size_t)255;
    return p;
  };

  const int S = S_LEN, HID = HIDDEN_, QR = Q_RANK_, KVR = KV_RANK_;
  const int H = NHEAD_;

  __bf16* hbf   = (__bf16*)alloc((size_t)S * HID * 2);
  __bf16* wqa   = (__bf16*)alloc((size_t)QR * HID * 2);
  __bf16* wqb   = (__bf16*)alloc((size_t)(H * D_QK_) * QR * 2);
  __bf16* wkva  = (__bf16*)alloc((size_t)(KVR + D_ROPE_) * HID * 2);
  __bf16* wkvb  = (__bf16*)alloc((size_t)(H * 256) * KVR * 2);
  __bf16* wo    = (__bf16*)alloc((size_t)HID * (H * D_V_) * 2);
  float*  qa    = (float*) alloc((size_t)S * QR * 4);
  __bf16* qaln  = (__bf16*)alloc((size_t)S * QR * 2);
  float*  qproj = (float*) alloc((size_t)S * (H * D_QK_) * 4);
  float*  kva   = (float*) alloc((size_t)S * (KVR + D_ROPE_) * 4);
  __bf16* ckv   = (__bf16*)alloc((size_t)S * KVR * 2);
  float*  kvb   = (float*) alloc((size_t)S * (H * 256) * 4);
  __bf16* qfull = (__bf16*)alloc((size_t)H * S * D_QK_ * 2);
  __bf16* kfull = (__bf16*)alloc((size_t)H * S * D_QK_ * 2);
  __bf16* vT    = (__bf16*)alloc((size_t)H * D_V_ * S * 2);
  __bf16* attn  = (__bf16*)alloc((size_t)S * (H * D_V_) * 2);

  auto cvt = [&](const float* x, __bf16* y, int n) {
    cvt_f32_bf16<<<(n + 255) / 256, 256, 0, stream>>>(x, y, n);
  };
  cvt(hidden, hbf,  S * HID);
  cvt(q_a_w,  wqa,  QR * HID);
  cvt(q_b_w,  wqb,  H * D_QK_ * QR);
  cvt(kv_a_w, wkva, (KVR + D_ROPE_) * HID);
  cvt(kv_b_w, wkvb, H * 256 * KVR);
  cvt(o_w,    wo,   HID * H * D_V_);

  // q_a = hidden @ q_a_w^T           [2048 x 1536], K=2048
  gemm_bf16_tn<<<dim3(S / 64, QR / 64), 128, 0, stream>>>(hbf, wqa, qa, S, QR, HID);
  // RMSNorm(q_a) -> bf16
  rmsnorm_bf16<<<S, 256, 0, stream>>>(qa, q_a_ln, qaln, QR, QR);
  // q = rms(q_a) @ q_b_w^T           [2048 x 3072], K=1536
  gemm_bf16_tn<<<dim3(S / 64, (H * D_QK_) / 64), 128, 0, stream>>>(
      qaln, wqb, qproj, S, H * D_QK_, QR);
  // kv_a = hidden @ kv_a_w^T         [2048 x 576], K=2048
  gemm_bf16_tn<<<dim3(S / 64, (KVR + D_ROPE_) / 64), 128, 0, stream>>>(
      hbf, wkva, kva, S, KVR + D_ROPE_, HID);
  // RMSNorm(kv_a[:, :512]) -> bf16
  rmsnorm_bf16<<<S, 256, 0, stream>>>(kva, kv_a_ln, ckv, KVR + D_ROPE_, KVR);
  // kv_b = ckv @ kv_b_w^T            [2048 x 4096], K=512
  gemm_bf16_tn<<<dim3(S / 64, (H * 256) / 64), 128, 0, stream>>>(
      ckv, wkvb, kvb, S, H * 256, KVR);
  // RoPE + per-head repack (+ V transpose)
  rope_pack<<<dim3(S, H), 128, 0, stream>>>(qproj, kvb, kva, pos, qfull, kfull, vT);
  // causal flash attention -> attn bf16 [S][H*128]
  mla_flash<<<dim3(S / 64, H), 128, 0, stream>>>(qfull, kfull, vT, attn);
  // out = attn @ o_w^T               [2048 x 2048], K=2048
  gemm_bf16_tn<<<dim3(S / 64, HID / 64), 128, 0, stream>>>(
      attn, wo, out, S, HID, H * D_V_);
}
